// GAT_skip_2_layer_sum_all_7567732376262
// MI455X (gfx1250) — matrix operand order repaired and checked by reading, not compiled, not run on hardware
//
#include <hip/hip_runtime.h>

// ---------------------------------------------------------------------------
// Types for CDNA5 WMMA (wave32): v_wmma_f32_16x16x32_bf16
// ---------------------------------------------------------------------------
typedef __attribute__((ext_vector_type(16))) __bf16 v16bf;
typedef __attribute__((ext_vector_type(8)))  float  v8f;

union Frag16 { v16bf v; unsigned int u[8]; uint4 q[2]; };

// ---------------------------------------------------------------------------
// Helpers
// ---------------------------------------------------------------------------
__device__ __forceinline__ unsigned short f32_to_bf16_rne(float f) {
    unsigned int u = __float_as_uint(f);
    unsigned int r = u + 0x7FFFu + ((u >> 16) & 1u);
    return (unsigned short)(r >> 16);
}

__device__ __forceinline__ float bf16_to_f32(unsigned short h) {
    return __uint_as_float(((unsigned int)h) << 16);
}

// Float atomic max via monotone integer punning (init value must be -inf).
__device__ __forceinline__ void atomicMaxF(float* addr, float val) {
    if (!(val < 0.0f)) {
        atomicMax((int*)addr, __float_as_int(val));
    } else {
        atomicMin((unsigned int*)addr, __float_as_uint(val));
    }
}

// ---------------------------------------------------------------------------
// f32 -> split bf16 (hi + lo) for error-compensated WMMA GEMM
// ---------------------------------------------------------------------------
__global__ void k_f32_to_bf16x2(const float* __restrict__ x,
                                unsigned short* __restrict__ hi,
                                unsigned short* __restrict__ lo,
                                long long n) {
    long long i = (long long)blockIdx.x * blockDim.x + threadIdx.x;
    long long stride = (long long)gridDim.x * blockDim.x;
    for (; i < n; i += stride) {
        float v = x[i];
        unsigned short h = f32_to_bf16_rne(v);
        unsigned short l = f32_to_bf16_rne(v - bf16_to_f32(h));
        hi[i] = h; lo[i] = l;
    }
}

// ---------------------------------------------------------------------------
// Pre-pack W (bf16 hi/lo) into WMMA B-fragment layout:
//   Bpack[((nt*4 + kc)*32 + lane)*8 + j] packs rows K = kc*32 + 8*(lane>=16) + 2*(j&3) + 16*(j>>2)
//   (lo half) and +1 (hi half), column N = nt*16 + (lane&15).
// One thread per (nt, kc, lane). Total = ntiles*128 threads.
// ---------------------------------------------------------------------------
__global__ void k_pack_bfrag(const unsigned short* __restrict__ Whi,
                             const unsigned short* __restrict__ Wlo,
                             unsigned int* __restrict__ Bhi,
                             unsigned int* __restrict__ Blo, int Ncols) {
    int idx = blockIdx.x * blockDim.x + threadIdx.x;
    int ntiles = Ncols >> 4;
    if (idx >= ntiles * 4 * 32) return;
    int lane = idx & 31;
    int kc   = (idx >> 5) & 3;
    int nt   = idx >> 7;
    int khalf = lane >> 4;
    int ncol  = (nt << 4) + (lane & 15);
    unsigned int* oh = Bhi + (size_t)idx * 8;
    unsigned int* ol = Blo + (size_t)idx * 8;
#pragma unroll
    for (int j = 0; j < 8; ++j) {
        int k0 = kc * 32 + 8 * khalf + 2 * (j & 3) + 16 * (j >> 2);
        oh[j] = (unsigned int)Whi[k0 * Ncols + ncol] |
                ((unsigned int)Whi[(k0 + 1) * Ncols + ncol] << 16);
        ol[j] = (unsigned int)Wlo[k0 * Ncols + ncol] |
                ((unsigned int)Wlo[(k0 + 1) * Ncols + ncol] << 16);
    }
}

// ---------------------------------------------------------------------------
// GEMM: C[M, NCOLS] = A[M,128] x W[128,NCOLS], near-fp32 via
// A_hi*W_hi + A_hi*W_lo + A_lo*W_hi, f32 accumulate in WMMA.
// One wave owns a 16-row strip. A frags: 2x b128 loads each; B frags: packed,
// 2x b128 loads each, L2-resident. K fixed at 128.
// ---------------------------------------------------------------------------
template <int NCOLS>
__global__ __launch_bounds__(256)
void k_gemm_bf16x2_wmma(const unsigned short* __restrict__ Ahi,
                        const unsigned short* __restrict__ Alo,
                        const unsigned int* __restrict__ Bhi,
                        const unsigned int* __restrict__ Blo,
                        float* __restrict__ C, int M) {
    const int K = 128;
    int wave = (int)(((long long)blockIdx.x * blockDim.x + threadIdx.x) >> 5);
    int nwaves = (M + 15) >> 4;
    if (wave >= nwaves) return;            // uniform per-wave branch, EXEC stays full
    int lane  = threadIdx.x & 31;
    int m0    = wave << 4;
    int khalf = lane >> 4;                 // 0: lanes 0-15, 1: lanes 16-31
    int row   = m0 + (lane & 15);
    if (row >= M) row = M - 1;             // clamp (keeps EXEC all-ones for WMMA)

    // A fragments (16x32 bf16 per k-chunk), ISA layout:
    //   lanes 0-15 : V0..V3 -> K=0..7,  V4..V7 -> K=16..23
    //   lanes 16-31: V0..V3 -> K=8..15, V4..V7 -> K=24..31
    // Per lane: dwords 0..3 = 16 contiguous bytes at row*K + kc*32 + 8*khalf,
    //           dwords 4..7 = 16 contiguous bytes at ... + 16.  (16B aligned)
    Frag16 ahi[4], alo[4];
    const unsigned short* arh = Ahi + (long long)row * K;
    const unsigned short* arl = Alo + (long long)row * K;
#pragma unroll
    for (int kc = 0; kc < 4; ++kc) {
        int o = kc * 32 + 8 * khalf;
        ahi[kc].q[0] = *(const uint4*)(arh + o);
        ahi[kc].q[1] = *(const uint4*)(arh + o + 16);
        alo[kc].q[0] = *(const uint4*)(arl + o);
        alo[kc].q[1] = *(const uint4*)(arl + o + 16);
    }

    constexpr int ntiles = NCOLS >> 4;
#pragma unroll
    for (int nt = 0; nt < ntiles; ++nt) {
        v8f acc = {};
#pragma unroll
        for (int kc = 0; kc < 4; ++kc) {
            const unsigned int* bb = Bhi + (((nt * 4 + kc) * 32 + lane) << 3);
            const unsigned int* bl = Blo + (((nt * 4 + kc) * 32 + lane) << 3);
            Frag16 fbh, fbl;
            fbh.q[0] = *(const uint4*)(bb);
            fbh.q[1] = *(const uint4*)(bb + 4);
            fbl.q[0] = *(const uint4*)(bl);
            fbl.q[1] = *(const uint4*)(bl + 4);
            acc = __builtin_amdgcn_wmma_f32_16x16x32_bf16(
                      false, ahi[kc].v, false, fbh.v, (short)0, acc, false, false);
            acc = __builtin_amdgcn_wmma_f32_16x16x32_bf16(
                      false, ahi[kc].v, false, fbl.v, (short)0, acc, false, false);
            acc = __builtin_amdgcn_wmma_f32_16x16x32_bf16(
                      false, alo[kc].v, false, fbh.v, (short)0, acc, false, false);
        }
        // C/D layout: VGPR r -> M=m0+r (lanes 0-15), M=m0+8+r (lanes 16-31)
#pragma unroll
        for (int r = 0; r < 8; ++r) {
            int rr = m0 + r + (khalf ? 8 : 0);
            if (rr < M) C[(long long)rr * NCOLS + (nt << 4) + (lane & 15)] = acc[r];
        }
    }
}

// ---------------------------------------------------------------------------
// Attention coefficients: alpha_src[n,h] = sum_c h[n,h,c]*a_src[h,c]
// ---------------------------------------------------------------------------
__global__ void k_alpha(const float* __restrict__ t,
                        const float* __restrict__ avs, const float* __restrict__ avd,
                        float* __restrict__ asrc, float* __restrict__ adst,
                        int Nn, int H, int Cc) {
    int idx = blockIdx.x * blockDim.x + threadIdx.x;
    if (idx >= Nn * H) return;
    int n = idx / H, h = idx - n * H;
    const float* rowp = t + (long long)n * H * Cc + h * Cc;
    float s1 = 0.f, s2 = 0.f;
    for (int c = 0; c < Cc; ++c) {
        float v = rowp[c];
        s1 += v * avs[h * Cc + c];
        s2 += v * avd[h * Cc + c];
    }
    asrc[idx] = s1; adst[idx] = s2;
}

// Pre-fill aggregation buffer with bias (F is a power of two: 64 or 128)
__global__ void k_fill_bias(float* __restrict__ out, const float* __restrict__ b,
                            long long n, int F) {
    long long i = (long long)blockIdx.x * blockDim.x + threadIdx.x;
    long long stride = (long long)gridDim.x * blockDim.x;
    for (; i < n; i += stride) out[i] = b[(int)(i & (F - 1))];
}

__global__ void k_init_mdenom(float* __restrict__ m, float* __restrict__ den, int n) {
    int i = blockIdx.x * blockDim.x + threadIdx.x;
    if (i < n) { m[i] = -__builtin_inff(); den[i] = 0.f; }
}

__global__ void k_zero(float* __restrict__ p, int n) {
    int i = blockIdx.x * blockDim.x + threadIdx.x;
    if (i < n) p[i] = 0.f;
}

// ---------------------------------------------------------------------------
// Edge passes. Edge e < Eraw comes from edge_index; e >= Eraw is self-loop.
// ---------------------------------------------------------------------------
__device__ __forceinline__ void edge_sd(const int* __restrict__ ei, int e, int Eraw,
                                        int& s, int& d) {
    if (e < Eraw) { s = ei[e]; d = ei[Eraw + e]; }
    else          { s = e - Eraw; d = s; }
}

__global__ void k_edge_max(const int* __restrict__ ei, int Eraw, int Etot,
                           const float* __restrict__ asrc, const float* __restrict__ adst,
                           float* __restrict__ ebuf, float* __restrict__ m, int logH) {
    int idx = blockIdx.x * blockDim.x + threadIdx.x;
    int total = Etot << logH;
    if (idx >= total) return;
    int e = idx >> logH, h = idx & ((1 << logH) - 1);
    int s, d; edge_sd(ei, e, Eraw, s, d);
    float v = asrc[(s << logH) + h] + adst[(d << logH) + h];
    v = (v > 0.f) ? v : 0.2f * v;             // leaky_relu, slope 0.2
    ebuf[idx] = v;
    atomicMaxF(m + ((d << logH) + h), v);
}

__global__ void k_edge_exp_sum(const int* __restrict__ ei, int Eraw, int Etot,
                               float* __restrict__ ebuf, const float* __restrict__ m,
                               float* __restrict__ den, int logH) {
    int idx = blockIdx.x * blockDim.x + threadIdx.x;
    int total = Etot << logH;
    if (idx >= total) return;
    int e = idx >> logH, h = idx & ((1 << logH) - 1);
    int s, d; edge_sd(ei, e, Eraw, s, d); (void)s;
    float p = __expf(ebuf[idx] - m[(d << logH) + h]);
    ebuf[idx] = p;
    atomicAdd(den + ((d << logH) + h), p);
}

// One wave per edge; lanes stride the F features; atomic scatter into L2.
__global__ __launch_bounds__(256)
void k_edge_scatter(const int* __restrict__ ei, int Eraw, int Etot,
                    const float* __restrict__ t, const float* __restrict__ p,
                    const float* __restrict__ den, float* __restrict__ agg,
                    int F, int logC, int logH) {
    int wv = (int)(((long long)blockIdx.x * blockDim.x + threadIdx.x) >> 5);
    if (wv >= Etot) return;
    int lane = threadIdx.x & 31;
    int s, d; edge_sd(ei, wv, Eraw, s, d);
    long long srow = (long long)s * F, drow = (long long)d * F;
    __builtin_prefetch(t + srow, 0, 0);       // global_prefetch of the gathered row
    for (int f = lane; f < F; f += 32) {
        int h = f >> logC;
        float alpha = p[((long long)wv << logH) + h] / den[(d << logH) + h];
        atomicAdd(agg + drow + f, alpha * t[srow + f]);
    }
}

// ---------------------------------------------------------------------------
// BatchNorm (training mode, biased variance), 128 channels.
// stats[0..127]=sum, stats[128..255]=sum of squares
// ---------------------------------------------------------------------------
__global__ void k_bn_stats(const float* __restrict__ x, float* __restrict__ stats, int Nn) {
    __shared__ float s[128], ss[128];
    int tid = threadIdx.x;
    if (tid < 128) { s[tid] = 0.f; ss[tid] = 0.f; }
    __syncthreads();
    int col = tid & 127;
    float ls = 0.f, lss = 0.f;
    for (long long r = (long long)blockIdx.x * 2 + (tid >> 7); r < Nn;
         r += (long long)gridDim.x * 2) {
        float v = x[r * 128 + col];
        ls += v; lss += v * v;
    }
    atomicAdd(&s[col], ls);
    atomicAdd(&ss[col], lss);
    __syncthreads();
    if (tid < 128) {
        atomicAdd(&stats[tid], s[tid]);
        atomicAdd(&stats[128 + tid], ss[tid]);
    }
}

__global__ void k_bn_relu(const float* __restrict__ x, const float* __restrict__ stats,
                          const float* __restrict__ g, const float* __restrict__ be,
                          float* __restrict__ y, int Nn) {
    long long n = (long long)Nn * 128;
    long long i = (long long)blockIdx.x * blockDim.x + threadIdx.x;
    long long stride = (long long)gridDim.x * blockDim.x;
    float invN = 1.0f / (float)Nn;
    for (; i < n; i += stride) {
        int col = (int)(i & 127);
        float mu  = stats[col] * invN;
        float var = stats[128 + col] * invN - mu * mu;
        float v = (x[i] - mu) * rsqrtf(var + 1e-5f) * g[col] + be[col];
        y[i] = (v > 0.f) ? v : 0.f;
    }
}

// hsum = out1 + relu(bn(x) + 0.5*hskip)
__global__ void k_bn_skip_sum(const float* __restrict__ x, const float* __restrict__ stats,
                              const float* __restrict__ g, const float* __restrict__ be,
                              const float* __restrict__ hskip, const float* __restrict__ out1,
                              float* __restrict__ hsum, int Nn) {
    long long n = (long long)Nn * 128;
    long long i = (long long)blockIdx.x * blockDim.x + threadIdx.x;
    long long stride = (long long)gridDim.x * blockDim.x;
    float invN = 1.0f / (float)Nn;
    for (; i < n; i += stride) {
        int col = (int)(i & 127);
        float mu  = stats[col] * invN;
        float var = stats[128 + col] * invN - mu * mu;
        float v = (x[i] - mu) * rsqrtf(var + 1e-5f) * g[col] + be[col] + 0.5f * hskip[i];
        v = (v > 0.f) ? v : 0.f;
        hsum[i] = out1[i] + v;
    }
}

// ---------------------------------------------------------------------------
// Host orchestration
// ---------------------------------------------------------------------------
extern "C" void kernel_launch(void* const* d_in, const int* in_sizes, int n_in,
                              void* d_out, int out_size, void* d_ws, size_t ws_size,
                              hipStream_t stream) {
    (void)n_in; (void)out_size; (void)ws_size;
    const float* x  = (const float*)d_in[0];
    const int*   ei = (const int*)d_in[1];       // [2, Eraw] flattened, per harness int
    const float* W[4]  = {(const float*)d_in[2],  (const float*)d_in[6],
                          (const float*)d_in[10], (const float*)d_in[14]};
    const float* As[4] = {(const float*)d_in[3],  (const float*)d_in[7],
                          (const float*)d_in[11], (const float*)d_in[15]};
    const float* Ad[4] = {(const float*)d_in[4],  (const float*)d_in[8],
                          (const float*)d_in[12], (const float*)d_in[16]};
    const float* Bi[4] = {(const float*)d_in[5],  (const float*)d_in[9],
                          (const float*)d_in[13], (const float*)d_in[17]};
    const float* g1  = (const float*)d_in[18]; const float* be1 = (const float*)d_in[19];
    const float* g2  = (const float*)d_in[20]; const float* be2 = (const float*)d_in[21];

    const int N    = in_sizes[0] / 128;
    const int Eraw = in_sizes[1] / 2;
    const int E    = Eraw + N;                   // + self loops

    // --- carve workspace ---
    size_t off = 0;
    auto carve = [&](size_t bytes) -> void* {
        void* p = (char*)d_ws + off;
        off += (bytes + 255) & ~(size_t)255;
        return p;
    };
    float* t    = (float*)carve((size_t)N * 128 * sizeof(float)); // GEMM out / later hsum
    float* agg  = (float*)carve((size_t)N * 128 * sizeof(float)); // aggregation
    float* h0   = (float*)carve((size_t)N * 128 * sizeof(float)); // skip branch
    float* out1 = (float*)carve((size_t)N * 128 * sizeof(float));
    unsigned short* Ahi = (unsigned short*)carve((size_t)N * 128 * 2);
    unsigned short* Alo = (unsigned short*)carve((size_t)N * 128 * 2);
    unsigned short* Whi = (unsigned short*)carve((size_t)128 * 128 * 2);
    unsigned short* Wlo = (unsigned short*)carve((size_t)128 * 128 * 2);
    unsigned int* Bph = (unsigned int*)carve((size_t)8 * 4 * 32 * 8 * 4); // packed B hi
    unsigned int* Bpl = (unsigned int*)carve((size_t)8 * 4 * 32 * 8 * 4); // packed B lo
    float* ebuf  = (float*)carve((size_t)E * 4 * sizeof(float));   // edge logits/probs
    float* asrc  = (float*)carve((size_t)N * 4 * sizeof(float));
    float* adst  = (float*)carve((size_t)N * 4 * sizeof(float));
    float* mbuf  = (float*)carve((size_t)N * 4 * sizeof(float));
    float* dbuf  = (float*)carve((size_t)N * 4 * sizeof(float));
    float* stats = (float*)carve(256 * sizeof(float));

    auto cdiv = [](long long a, long long b) { return (int)((a + b - 1) / b); };

    // One full GAT layer: tout = feat@W, agg_out = segment_softmax_agg + bias
    auto run_gat = [&](const float* feat, const float* Wl, const float* avs,
                       const float* avd, const float* bl, int H, int Cc, int logH,
                       int logC, float* tout, float* aggout) {
        int F = H * Cc;
        k_f32_to_bf16x2<<<4096, 256, 0, stream>>>(feat, Ahi, Alo, (long long)N * 128);
        k_f32_to_bf16x2<<<64, 256, 0, stream>>>(Wl, Whi, Wlo, (long long)128 * F);
        int ntiles = F >> 4;
        k_pack_bfrag<<<cdiv((long long)ntiles * 128, 256), 256, 0, stream>>>(
            Whi, Wlo, Bph, Bpl, F);
        int waves = (N + 15) / 16;
        int gblk = cdiv((long long)waves * 32, 256);
        if (F == 128)
            k_gemm_bf16x2_wmma<128><<<gblk, 256, 0, stream>>>(Ahi, Alo, Bph, Bpl, tout, N);
        else
            k_gemm_bf16x2_wmma<64><<<gblk, 256, 0, stream>>>(Ahi, Alo, Bph, Bpl, tout, N);
        k_alpha<<<cdiv((long long)N * H, 256), 256, 0, stream>>>(
            tout, avs, avd, asrc, adst, N, H, Cc);
        k_fill_bias<<<2048, 256, 0, stream>>>(aggout, bl, (long long)N * F, F);
        k_init_mdenom<<<cdiv((long long)N * H, 256), 256, 0, stream>>>(mbuf, dbuf, N * H);
        k_edge_max<<<cdiv((long long)E * H, 256), 256, 0, stream>>>(
            ei, Eraw, E, asrc, adst, ebuf, mbuf, logH);
        k_edge_exp_sum<<<cdiv((long long)E * H, 256), 256, 0, stream>>>(
            ei, Eraw, E, ebuf, mbuf, dbuf, logH);
        k_edge_scatter<<<cdiv((long long)E * 32, 256), 256, 0, stream>>>(
            ei, Eraw, E, tout, ebuf, dbuf, aggout, F, logC, logH);
    };

    // Layer 0: x -> h0 (skip source), 4 heads x 32
    run_gat(x, W[0], As[0], Ad[0], Bi[0], 4, 32, 2, 5, t, h0);

    // Layer 1: h0 -> agg; out1 = relu(bn1(agg))
    run_gat(h0, W[1], As[1], Ad[1], Bi[1], 4, 32, 2, 5, t, agg);
    k_zero<<<1, 256, 0, stream>>>(stats, 256);
    k_bn_stats<<<1024, 256, 0, stream>>>(agg, stats, N);
    k_bn_relu<<<4096, 256, 0, stream>>>(agg, stats, g1, be1, out1, N);

    // Layer 2: out1 -> agg; hsum(t) = out1 + relu(bn2(agg) + 0.5*h0)
    run_gat(out1, W[2], As[2], Ad[2], Bi[2], 4, 32, 2, 5, t, agg);
    k_zero<<<1, 256, 0, stream>>>(stats, 256);
    k_bn_stats<<<1024, 256, 0, stream>>>(agg, stats, N);
    k_bn_skip_sum<<<4096, 256, 0, stream>>>(agg, stats, g2, be2, h0, out1, t, N);

    // Layer 3: hsum(t) -> d_out, 1 head x 64 (convert reads t before GEMM rewrites it)
    run_gat(t, W[3], As[3], Ad[3], Bi[3], 1, 64, 0, 6, t, (float*)d_out);
}
